// MultiHeadAttention_10187662426546
// MI455X (gfx1250) — matrix operand order repaired and checked by reading, not compiled, not run on hardware
//
#include <hip/hip_runtime.h>

// MHA forward for MI455X (gfx1250): wave32, bf16 WMMA 16x16x32, f32 accum.
// Attention K/V tiles staged to LDS by the Tensor Data Mover (TDM),
// double-buffered, synchronized with s_wait_tensorcnt + workgroup barriers.
// B=2, S=2048, D=1024, H=16, DH=64.

#define BB 2
#define SS 2048
#define DD 1024
#define HH 16
#define DH 64

typedef __attribute__((ext_vector_type(16))) __bf16 v16bf;
typedef __attribute__((ext_vector_type(8)))  __bf16 v8bf;
typedef __attribute__((ext_vector_type(8)))  float  v8f;
typedef __attribute__((ext_vector_type(8)))  float  float8;
typedef __attribute__((ext_vector_type(4)))  unsigned int uint32x4;
typedef __attribute__((ext_vector_type(4)))  int    int32x4;
typedef __attribute__((ext_vector_type(8)))  int    int32x8;

// LDS tile pitch: 64 bf16 payload + 8 bf16 TDM pad (4 DWORDs) = 144 bytes.
#define LPITCH 72

__device__ __forceinline__ v8f wmma_bf16(v16bf a, v16bf b, v8f c) {
  // D = A(16x32) * B(32x16) + C, f32 accumulation
  return __builtin_amdgcn_wmma_f32_16x16x32_bf16(false, a, false, b, (short)0, c,
                                                 false, false);
}

// A-operand (16-bit): lane row m = lane&15,
// elems 0..15 <-> K = 16*(lane>>4)+j (contiguous, matches B).
__device__ __forceinline__ v16bf load_a_bf16(const __bf16* p, int kbase) {
  v8bf lo = *(const v8bf*)(p + 2 * kbase);
  v8bf hi = *(const v8bf*)(p + 2 * kbase + 8);
  return __builtin_shufflevector(lo, hi, 0, 1, 2, 3, 4, 5, 6, 7,
                                 8, 9, 10, 11, 12, 13, 14, 15);
}

__device__ __forceinline__ v16bf load_a_f32(const float* p, int kbase) {
  float8 lo = *(const float8*)(p + 2 * kbase);
  float8 hi = *(const float8*)(p + 2 * kbase + 8);
  v8bf l = __builtin_convertvector(lo, v8bf);
  v8bf h = __builtin_convertvector(hi, v8bf);
  return __builtin_shufflevector(l, h, 0, 1, 2, 3, 4, 5, 6, 7,
                                 8, 9, 10, 11, 12, 13, 14, 15);
}

// B-operand (16-bit): lane col n = lane&15, elems 0..15 = 16 contiguous K
// starting at 16*(lane>>4). One 32-byte load.
__device__ __forceinline__ v16bf load_b_bf16(const __bf16* p, int khalf) {
  return *(const v16bf*)(p + 16 * khalf);
}

// ---------------------------------------------------------------------------
// TDM: 2-D tile load Global -> LDS (ISA 08_async_tensor §8).
// D# group0: count=1 | lds_addr | global_addr(57b) | type=2.
// D# group1: mask=0, data_size=1 (2B), pad_enable, pad_interval=4 (32 DWORDs
//            = one 64-elem row), pad_amount=3 (4 DWORDs) -> 144B LDS pitch,
//            tensor_dim0/1, tile_dim0/1, tensor_dim0_stride.
// Groups 2/3 zero (2-D tile: tile_dim2 = 0).
// ---------------------------------------------------------------------------
__device__ __forceinline__ void tdm_load_2d(unsigned lds_off, const void* gaddr,
                                            unsigned tensor_d0, unsigned tensor_d1,
                                            unsigned tile_d0, unsigned tile_d1,
                                            unsigned stride0_elems) {
  unsigned long long ga = (unsigned long long)(uintptr_t)gaddr;
  uint32x4 g0;
  g0[0] = 1u;                                   // count=1, user descriptor
  g0[1] = lds_off;                              // LDS byte address
  g0[2] = (unsigned)(ga & 0xffffffffu);
  g0[3] = (unsigned)((ga >> 32) & 0x1ffffffu) | (2u << 30);  // type=2 (image)
  int32x8 g1;
  g1[0] = (int)((1u << 16) | (1u << 20) | (4u << 22) | (3u << 25));
  g1[1] = (int)((tensor_d0 & 0xffffu) << 16);
  g1[2] = (int)(((tensor_d0 >> 16) & 0xffffu) | ((tensor_d1 & 0xffffu) << 16));
  g1[3] = (int)(((tensor_d1 >> 16) & 0xffffu) | ((tile_d0 & 0xffffu) << 16));
  g1[4] = (int)(tile_d1 & 0xffffu);
  g1[5] = (int)stride0_elems;                   // tensor_dim0_stride[31:0]
  g1[6] = 0;                                    // stride0[47:32], stride1[15:0]
  g1[7] = 0;
  int32x4 z4 = {0, 0, 0, 0};
#if defined(__clang_major__) && (__clang_major__ >= 23)
  int32x8 z8 = {0, 0, 0, 0, 0, 0, 0, 0};
  __builtin_amdgcn_tensor_load_to_lds(g0, g1, z4, z4, z8, 0);
#else
  __builtin_amdgcn_tensor_load_to_lds(g0, g1, z4, z4, 0);
#endif
}

// ---------------------------------------------------------------------------
// Kernel 0: W[k][n] f32 -> Wt[n][k] bf16 for Wq,Wk,Wv,Wo (contiguous mats).
// ---------------------------------------------------------------------------
__global__ void wt_kernel(const float* __restrict__ Wq, const float* __restrict__ Wk,
                          const float* __restrict__ Wv, const float* __restrict__ Wo,
                          __bf16* __restrict__ Wt) {
  int t = blockIdx.x * blockDim.x + threadIdx.x;   // 0 .. 4*1024*1024-1
  int mat = t >> 20;
  int rem = t & ((1 << 20) - 1);
  int n = rem >> 10;
  int k = rem & 1023;
  const float* src = (mat == 0) ? Wq : (mat == 1) ? Wk : (mat == 2) ? Wv : Wo;
  Wt[(size_t)t] = (__bf16)src[(size_t)k * DD + n];   // Wt[mat][n][k]
}

// ---------------------------------------------------------------------------
// Kernel 1: fused QKV projection.  grid=(BS/64, 3*16), block=128 (4 waves).
// Q pre-scaled by 0.125*log2(e) (softmax in exp2 domain).
// Q,K stored [B,H,S,DH] bf16; V stored transposed [B,H,DH,S] bf16.
// ---------------------------------------------------------------------------
__global__ __launch_bounds__(128) void qkv_kernel(
    const float* __restrict__ x,
    const float* __restrict__ bq, const float* __restrict__ bk,
    const float* __restrict__ bv,
    const __bf16* __restrict__ Wt,
    __bf16* __restrict__ Qb, __bf16* __restrict__ Kb, __bf16* __restrict__ Vt) {
  const int lane = threadIdx.x & 31;
  const int wave = threadIdx.x >> 5;
  const int l16 = lane & 15;
  const int khalf = lane >> 4;
  const int kbase = khalf * 8;

  const int mat = blockIdx.y >> 4;            // 0=Q 1=K 2=V
  const int c0 = (blockIdx.y & 15) * 64;      // column block within D
  const int s0 = blockIdx.x * 64 + wave * 16; // token rows for this wave

  const float* xrow = x + (size_t)(s0 + l16) * DD;
  const __bf16* wbase = Wt + (size_t)mat * DD * DD;

  v8f acc[4] = {};
  for (int k0 = 0; k0 < DD; k0 += 32) {
    v16bf a = load_a_f32(xrow + k0, kbase);
#pragma unroll
    for (int g = 0; g < 4; ++g) {
      v16bf b = load_b_bf16(wbase + (size_t)(c0 + 16 * g + l16) * DD + k0, khalf);
      acc[g] = wmma_bf16(a, b, acc[g]);
    }
  }

  const float* bias = (mat == 0) ? bq : (mat == 1) ? bk : bv;
  const float qs = 0.125f * 1.4426950408889634f;

#pragma unroll
  for (int g = 0; g < 4; ++g) {
    const int c = c0 + 16 * g + l16;
    const int h = c >> 6, dh = c & 63;
    const float bia = bias[c];
    if (mat == 2) {
      const int sg0 = s0 + 8 * khalf;
      const int bidx = sg0 >> 11, sl = sg0 & (SS - 1);
      v8bf pk;
#pragma unroll
      for (int r = 0; r < 8; ++r) pk[r] = (__bf16)(acc[g][r] + bia);
      *(v8bf*)(Vt + (((size_t)bidx * HH + h) * DH + dh) * SS + sl) = pk;
    } else {
      __bf16* dst = (mat == 0) ? Qb : Kb;
#pragma unroll
      for (int r = 0; r < 8; ++r) {
        const int sg = s0 + 8 * khalf + r;
        const int bidx = sg >> 11, sl = sg & (SS - 1);
        float v = acc[g][r] + bia;
        if (mat == 0) v *= qs;
        dst[(((size_t)bidx * HH + h) * SS + sl) * DH + dh] = (__bf16)v;
      }
    }
  }
}

// ---------------------------------------------------------------------------
// Kernel 2: flash attention, transposed scores, TDM-staged K/V tiles.
// grid=(S/64, B*H), block=128 (4 waves, 64 queries). Key blocks of 64:
// wave 0 issues tensor_load_to_lds for K(64x64) and V^T(64x64) into the
// back buffer, s_wait_tensorcnt(2) retires the front buffer, barriers
// publish it; all waves read A-operands from LDS (144B pitch, conflict-free).
// St(16k x 16q) = K_tile @ Q^T ; O^T(64d x 16q) += V^T_tile @ P^T where the
// P^T B-operand is built in-register from the exp'd score tiles with one
// shfl_xor(16) per element pair.
// ---------------------------------------------------------------------------
__global__ __launch_bounds__(128) void attn_kernel(
    const __bf16* __restrict__ Qb, const __bf16* __restrict__ Kb,
    const __bf16* __restrict__ Vt, __bf16* __restrict__ Obuf) {
  __shared__ __align__(16) __bf16 kbuf[2][64 * LPITCH];  // 2 x 9 KB
  __shared__ __align__(16) __bf16 vbuf[2][64 * LPITCH];  // 2 x 9 KB

  const int lane = threadIdx.x & 31;
  const int wave = threadIdx.x >> 5;
  const int l16 = lane & 15;
  const int khalf = lane >> 4;
  const int kbase = khalf * 8;

  const int bh = blockIdx.y;                  // b*16 + h
  const int q0 = blockIdx.x * 64 + wave * 16;

  // Q^T B-operands (held for the whole kernel): lane col = query q0+l16
  const __bf16* qrow = Qb + ((size_t)bh * SS + q0 + l16) * DH;
  const v16bf qop0 = load_b_bf16(qrow, khalf);        // d = 0..31
  const v16bf qop1 = load_b_bf16(qrow + 32, khalf);   // d = 32..63

  const __bf16* kgbase = Kb + (size_t)bh * SS * DH;   // [S][64] row-major
  const __bf16* vgbase = Vt + (size_t)bh * DH * SS;   // [64][S] row-major

  v8f o[4] = {};                 // O^T accumulator: 64 d-rows x 16 q-cols
  float m = -1e30f, ssum = 0.0f; // per-query online softmax stats (q = l16)

  const int nblk = SS / 64;

  if (wave == 0) {
    // Prologue: DMA block 0 into buffer 0.
    tdm_load_2d((unsigned)(uintptr_t)&kbuf[0][0], kgbase, DH, SS, 64, 64, DH);
    tdm_load_2d((unsigned)(uintptr_t)&vbuf[0][0], vgbase, SS, DH, 64, 64, SS);
  }

  for (int blk = 0; blk < nblk; ++blk) {
    const int buf = blk & 1;
    if (wave == 0) {
      if (blk + 1 < nblk) {
        const int tn = (blk + 1) * 64;
        tdm_load_2d((unsigned)(uintptr_t)&kbuf[buf ^ 1][0],
                    kgbase + (size_t)tn * DH, DH, SS, 64, 64, DH);
        tdm_load_2d((unsigned)(uintptr_t)&vbuf[buf ^ 1][0],
                    vgbase + tn, SS, DH, 64, 64, SS);
        __builtin_amdgcn_s_wait_tensorcnt(2);  // front buffer's 2 ops retired
      } else {
        __builtin_amdgcn_s_wait_tensorcnt(0);
      }
    }
    __syncthreads();   // publish front buffer to all 4 waves

    const __bf16* kb = &kbuf[buf][0];
    const __bf16* vb = &vbuf[buf][0];

#pragma unroll
    for (int sub = 0; sub < 64; sub += 32) {
      const __bf16* kr0 = kb + (sub + l16) * LPITCH;
      const __bf16* kr1 = kb + (sub + 16 + l16) * LPITCH;
      v16bf ka00 = load_a_bf16(kr0, kbase);        // keys sub..sub+15, d 0..31
      v16bf ka01 = load_a_bf16(kr0 + 32, kbase);   //                  d 32..63
      v16bf ka10 = load_a_bf16(kr1, kbase);        // keys sub+16..sub+31
      v16bf ka11 = load_a_bf16(kr1 + 32, kbase);

      v8f sc0 = {}, sc1 = {};
      sc0 = wmma_bf16(ka00, qop0, sc0);
      sc0 = wmma_bf16(ka01, qop1, sc0);
      sc1 = wmma_bf16(ka10, qop0, sc1);
      sc1 = wmma_bf16(ka11, qop1, sc1);

      // per-query tile max: 16 values intra-lane + pair reduction over lane^16
      float tm = -1e30f;
#pragma unroll
      for (int r = 0; r < 8; ++r) {
        tm = fmaxf(tm, sc0[r]);
        tm = fmaxf(tm, sc1[r]);
      }
      tm = fmaxf(tm, __shfl_xor(tm, 16, 32));
      const float mnew = fmaxf(m, tm);
      const float f = __builtin_amdgcn_exp2f(m - mnew);

      // exp + build P^T B-operand: elem j <-> t_local = 16*khalf + j,
      // elem 8+j <-> t_local = 16*khalf + 8 + j. Own/partner select per half.
      v16bf pb;
      float ps = 0.0f;
#pragma unroll
      for (int r = 0; r < 8; ++r) {
        const float p0 = __builtin_amdgcn_exp2f(sc0[r] - mnew);
        const float p1 = __builtin_amdgcn_exp2f(sc1[r] - mnew);
        ps += p0 + p1;
        union { unsigned u; __bf16 h[2]; } own, oth;
        own.h[0] = (__bf16)p0;
        own.h[1] = (__bf16)p1;
        oth.u = (unsigned)__shfl_xor((int)own.u, 16, 32);
        pb[r]     = khalf ? oth.h[1] : own.h[0];
        pb[8 + r] = khalf ? own.h[1] : oth.h[0];
      }
      ps += __shfl_xor(ps, 16, 32);
      ssum = ssum * f + ps;

#pragma unroll
      for (int g = 0; g < 4; ++g) {
#pragma unroll
        for (int r = 0; r < 8; ++r) o[g][r] *= f;
        // V^T A-operand from LDS: row d = 16g + l16, k = t (contiguous)
        v16bf va = load_a_bf16(vb + (16 * g + l16) * LPITCH + sub, kbase);
        o[g] = wmma_bf16(va, pb, o[g]);
      }
      m = mnew;
    }
    __syncthreads();   // all waves done reading before buffer is overwritten
  }

  const float inv = 1.0f / ssum;
  const int b = bh >> 4, h = bh & 15;
#pragma unroll
  for (int g = 0; g < 4; ++g) {
    v8bf pk;
#pragma unroll
    for (int r = 0; r < 8; ++r) pk[r] = (__bf16)(o[g][r] * inv);
    // lane: query fixed (=l16), 8 consecutive d -> one b128 store
    *(v8bf*)(Obuf + (size_t)(b * SS + q0 + l16) * DD + h * DH + 16 * g + 8 * khalf) = pk;
  }
}

// ---------------------------------------------------------------------------
// Kernel 3: output projection Obuf[4096,1024]bf16 @ Wo^T + bo -> f32 out.
// ---------------------------------------------------------------------------
__global__ __launch_bounds__(128) void out_kernel(
    const __bf16* __restrict__ Obuf, const __bf16* __restrict__ WoT,
    const float* __restrict__ bo, float* __restrict__ out) {
  const int lane = threadIdx.x & 31;
  const int wave = threadIdx.x >> 5;
  const int l16 = lane & 15;
  const int khalf = lane >> 4;
  const int kbase = khalf * 8;

  const int s0 = blockIdx.x * 64 + wave * 16;
  const int c0 = blockIdx.y * 64;

  const __bf16* arow = Obuf + (size_t)(s0 + l16) * DD;

  v8f acc[4] = {};
  for (int k0 = 0; k0 < DD; k0 += 32) {
    v16bf a = load_a_bf16(arow + k0, kbase);
#pragma unroll
    for (int g = 0; g < 4; ++g) {
      v16bf b = load_b_bf16(WoT + (size_t)(c0 + 16 * g + l16) * DD + k0, khalf);
      acc[g] = wmma_bf16(a, b, acc[g]);
    }
  }

#pragma unroll
  for (int g = 0; g < 4; ++g) {
    const int c = c0 + 16 * g + l16;
    const float bia = bo[c];
#pragma unroll
    for (int r = 0; r < 8; ++r) {
      const int s = s0 + 8 * khalf + r;
      out[(size_t)s * DD + c] = acc[g][r] + bia;
    }
  }
}

// ---------------------------------------------------------------------------
extern "C" void kernel_launch(void* const* d_in, const int* in_sizes, int n_in,
                              void* d_out, int out_size, void* d_ws, size_t ws_size,
                              hipStream_t stream) {
  (void)in_sizes; (void)n_in; (void)out_size; (void)ws_size;
  const float* x  = (const float*)d_in[0];
  const float* Wq = (const float*)d_in[1];
  const float* bq = (const float*)d_in[2];
  const float* Wk = (const float*)d_in[3];
  const float* bk = (const float*)d_in[4];
  const float* Wv = (const float*)d_in[5];
  const float* bv = (const float*)d_in[6];
  const float* Wo = (const float*)d_in[7];
  const float* bo = (const float*)d_in[8];
  float* out = (float*)d_out;

  // Workspace layout (40 MB total):
  //  [0,8MB)   Wt: 4x [1024][1024] bf16 (Wq^T, Wk^T, Wv^T, Wo^T)
  //  [8,16MB)  Qb  [B,H,S,64]  bf16 (pre-scaled by 0.125*log2e)
  //  [16,24MB) Kb  [B,H,S,64]  bf16
  //  [24,32MB) Vt  [B,H,64,S]  bf16 (transposed V)
  //  [32,40MB) Obuf[B*S,1024]  bf16 (attention output, pre-projection)
  char* ws = (char*)d_ws;
  __bf16* Wt   = (__bf16*)(ws);
  __bf16* Qb   = (__bf16*)(ws + ((size_t)8 << 20));
  __bf16* Kb   = (__bf16*)(ws + ((size_t)16 << 20));
  __bf16* Vt   = (__bf16*)(ws + ((size_t)24 << 20));
  __bf16* Obuf = (__bf16*)(ws + ((size_t)32 << 20));
  __bf16* WoT  = Wt + (size_t)3 * DD * DD;

  hipLaunchKernelGGL(wt_kernel, dim3(16384), dim3(256), 0, stream,
                     Wq, Wk, Wv, Wo, Wt);
  hipLaunchKernelGGL(qkv_kernel, dim3(64, 48), dim3(128), 0, stream,
                     x, bq, bk, bv, Wt, Qb, Kb, Vt);
  hipLaunchKernelGGL(attn_kernel, dim3(32, 32), dim3(128), 0, stream,
                     Qb, Kb, Vt, Obuf);
  hipLaunchKernelGGL(out_kernel, dim3(64, 16), dim3(128), 0, stream,
                     Obuf, WoT, bo, out);
}
